// CTMHierModel_57423712748045
// MI455X (gfx1250) — compile-verified, matrix operand order checked by reading, not run on hardware
//
#include <hip/hip_runtime.h>
#include <hip/hip_bf16.h>
#include <math.h>

// ---------------------------------------------------------------------------
// CTM hierarchical model on gfx1250 (MI455X), WMMA f16->f32 path.
//
// Layout: each wave owns 16 batch elements -> 80 activation rows (16 el x 5
// tokens) of 32 f16 dims in LDS.  Every 32x32 linear = 5 row-tiles x 2
// N-halves of v_wmma_f32_16x16x32_f16.  Weights pre-converted to transposed
// f16 (W^T, 32 rows x 32 K, K zero-padded for fin<32) in d_ws by a prep
// kernel, so B fragments load with two b128 loads exactly like A fragments.
// ---------------------------------------------------------------------------

typedef __attribute__((ext_vector_type(16))) _Float16 v16h;
typedef __attribute__((ext_vector_type(8)))  _Float16 v8h;
typedef __attribute__((ext_vector_type(8)))  float    v8f;

#define BATCH      1048576
#define ELEM_PER_WAVE 16
#define ROWS       80            // 16 elements * 5 tokens
#define WAVES_PER_BLOCK 2
#define SCALAR_BASE ((size_t)BATCH * 4)   // out[SCALAR_BASE]=pred_loss, +1=eff_p

// ---------------- fragment helpers (wave32 WMMA 16x16x32 f16 layouts) -------

// A / B fragment from a row-major 32-halves-per-row tile.
// 16-bit A 16x32 layout: lanes 0-15 hold K 0..7 & 16..23, lanes 16-31 hold
// K 8..15 & 24..31 of row (lane&15).  B uses the same pattern over W^T rows.
__device__ __forceinline__ v16h frag_from_rows(const _Float16* base, int row) {
  const int lane = threadIdx.x & 31;
  const _Float16* p = base + row * 32 + ((lane >> 4) << 3);
  v8h lo = *(const v8h*)(p);        // K = kb .. kb+7
  v8h hi = *(const v8h*)(p + 16);   // K = 16+kb .. 16+kb+7
  v16h f;
#pragma unroll
  for (int i = 0; i < 8; ++i) { f[i] = lo[i]; f[8 + i] = hi[i]; }
  return f;
}

// Scatter a D accumulator (f32 16x16) into a row-major f16 tile.
// Lanes 0-15: M = v, lanes 16-31: M = v+8; N = lane&15 (+16 for nh=1).
__device__ __forceinline__ void store_d(_Float16* base, int row0, int nh,
                                        v8f d, bool dorelu) {
  const int lane = threadIdx.x & 31;
  const int col   = nh * 16 + (lane & 15);
  const int rbase = row0 + ((lane >> 4) << 3);
#pragma unroll
  for (int v = 0; v < 8; ++v) {
    float x = d[v];
    if (dorelu) x = x > 0.f ? x : 0.f;
    base[(rbase + v) * 32 + col] = (_Float16)x;
  }
}

__device__ __forceinline__ v8f bias_acc(const float* bias, int nh) {
  const int lane = threadIdx.x & 31;
  float b = bias[nh * 16 + (lane & 15)];
  v8f c;
#pragma unroll
  for (int v = 0; v < 8; ++v) c[v] = b;
  return c;
}

// Y(80x32) = act( X(80x32) @ W(32x32) + b ) : 10 WMMAs.
__device__ __forceinline__ void lin5(const _Float16* src, _Float16* dst,
                                     const _Float16* wt, const float* bias,
                                     bool dorelu) {
  const int lane = threadIdx.x & 31;
  const int n = lane & 15;
  v16h b0 = frag_from_rows(wt, n);        // N-half 0 of W^T
  v16h b1 = frag_from_rows(wt, 16 + n);   // N-half 1
#pragma unroll
  for (int rt = 0; rt < 5; ++rt) {
    v16h a = frag_from_rows(src, rt * 16 + n);
    v8f c0 = bias_acc(bias, 0);
    v8f c1 = bias_acc(bias, 1);
    c0 = __builtin_amdgcn_wmma_f32_16x16x32_f16(false, a, false, b0, (short)0, c0, false, false);
    c1 = __builtin_amdgcn_wmma_f32_16x16x32_f16(false, a, false, b1, (short)0, c1, false, false);
    store_d(dst, rt * 16, 0, c0, dorelu);
    store_d(dst, rt * 16, 1, c1, dorelu);
  }
}

// Same, but row-tiles 0..3 (obj rows) use weight A, row-tile 4 (query rows)
// uses weight B.  Used by the two embedding layers.
__device__ __forceinline__ void lin5_mix(const _Float16* src, _Float16* dst,
                                         const _Float16* wtA, const float* bA,
                                         const _Float16* wtB, const float* bB,
                                         bool dorelu) {
  const int lane = threadIdx.x & 31;
  const int n = lane & 15;
  v16h a0 = frag_from_rows(wtA, n),      a1 = frag_from_rows(wtA, 16 + n);
  v16h q0 = frag_from_rows(wtB, n),      q1 = frag_from_rows(wtB, 16 + n);
#pragma unroll
  for (int rt = 0; rt < 5; ++rt) {
    const bool obj = rt < 4;
    v16h w0 = obj ? a0 : q0;
    v16h w1 = obj ? a1 : q1;
    const float* bb = obj ? bA : bB;
    v16h a = frag_from_rows(src, rt * 16 + n);
    v8f c0 = bias_acc(bb, 0);
    v8f c1 = bias_acc(bb, 1);
    c0 = __builtin_amdgcn_wmma_f32_16x16x32_f16(false, a, false, w0, (short)0, c0, false, false);
    c1 = __builtin_amdgcn_wmma_f32_16x16x32_f16(false, a, false, w1, (short)0, c1, false, false);
    store_d(dst, rt * 16, 0, c0, dorelu);
    store_d(dst, rt * 16, 1, c1, dorelu);
  }
}

// ---------------- parameter plumbing ---------------------------------------

struct PrepP { const float* w[18]; };

// Convert all 18 weight matrices (fin x 32, f32, row-major) into transposed
// f16 tiles (32 N-rows x 32 K-cols, K zero-padded) in d_ws.
__global__ void prep_weights(PrepP P, _Float16* wt) {
  int gid = blockIdx.x * 256 + threadIdx.x;
  if (gid >= 18 * 1024) return;
  const int finTab[18] = {10, 32, 12, 32,  32, 32, 32, 32, 32, 32, 32,
                          32, 32, 32, 32, 32, 32, 32};
  int m  = gid >> 10;
  int nn = (gid >> 5) & 31;   // output (N) index -> row of W^T
  int kk = gid & 31;          // input  (K) index -> col of W^T
  float v = (kk < finTab[m]) ? P.w[m][kk * 32 + nn] : 0.0f;
  wt[gid] = (_Float16)v;
}

__global__ void init_out(float* out) {
  if (threadIdx.x == 0) { out[SCALAR_BASE] = 0.f; out[SCALAR_BASE + 1] = 0.f; }
}

struct MainP {
  const float* x;
  float*       out;
  const float* eb[4];        // se1_b, se2_b, qe1_b, qe2_b
  const float* bb[2][7];     // per CTM block: in,q,k,v,o,p1,p2 biases
  const float* ln_g[2];
  const float* ln_b[2];
  const float* ltau[2];
  const float* head_w;
  const float* head_b;
};

// ---------------- main kernel ----------------------------------------------

__global__ __launch_bounds__(64) void ctm_kernel(MainP P, const _Float16* __restrict__ wt) {
  __shared__ _Float16 sZ[WAVES_PER_BLOCK][ROWS * 32];
  __shared__ _Float16 sH[WAVES_PER_BLOCK][ROWS * 32];
  __shared__ _Float16 sQ[WAVES_PER_BLOCK][ROWS * 32];
  __shared__ _Float16 sK[WAVES_PER_BLOCK][ROWS * 32];
  __shared__ _Float16 sV[WAVES_PER_BLOCK][ROWS * 32];
  __shared__ _Float16 sP[WAVES_PER_BLOCK][ROWS * 32];
  __shared__ float    sSim[WAVES_PER_BLOCK][16 * 30]; // 0..24 sim/attn, 25..29 aux
  __shared__ _Float16 sItau[2][32];                   // 1/(exp(log_tau)+1e-6)

  const int wid  = threadIdx.x >> 5;
  const int lane = threadIdx.x & 31;
  const size_t eBase = ((size_t)blockIdx.x * WAVES_PER_BLOCK + wid) * ELEM_PER_WAVE;

  _Float16* Z = sZ[wid];  _Float16* H = sH[wid];
  _Float16* Q = sQ[wid];  _Float16* K = sK[wid];
  _Float16* V = sV[wid];  _Float16* Pr = sP[wid];
  float* SIM = sSim[wid];

  const _Float16* wSe1 = wt + 0 * 1024;
  const _Float16* wSe2 = wt + 1 * 1024;
  const _Float16* wQe1 = wt + 2 * 1024;
  const _Float16* wQe2 = wt + 3 * 1024;

  float plAcc = 0.f, epAcc = 0.f;

  // precompute per-dim 1/(tau+eps): one entry per (block, dim), shared by
  // both waves of the workgroup (64 threads cover the 64 entries).
  {
    int b = threadIdx.x >> 5, d = threadIdx.x & 31;
    sItau[b][d] = (_Float16)(1.0f / (expf(P.ltau[b][d]) + 1e-6f));
  }

  // ---- stage inputs: obj rows (K-padded 10->32) then q rows (12->32) ------
  // staging layout in H: rows 0..63 = obj rows (4 per element), 64..79 = q.
  for (int r = lane; r < ROWS; r += 32) {
    int e, off, cnt;
    if (r < 64) { e = r >> 2; off = (r & 3) * 10; cnt = 10; }
    else        { e = r - 64; off = 40;          cnt = 12; }
    const float* xp = P.x + (eBase + e) * 52 + off;
    __builtin_prefetch(xp + ELEM_PER_WAVE * WAVES_PER_BLOCK * 52, 0, 1);
    _Float16* row = &H[r * 32];
#pragma unroll 4
    for (int d = 0; d < 32; ++d) row[d] = (_Float16)(d < cnt ? xp[d] : 0.0f);
  }
  __syncthreads();

  // ---- embeddings ---------------------------------------------------------
  lin5_mix(H, Q, wSe1, P.eb[0], wQe1, P.eb[2], /*relu*/true);
  __syncthreads();
  lin5_mix(Q, K, wSe2, P.eb[1], wQe2, P.eb[3], /*relu*/false);
  __syncthreads();

  // repack staged rows into z layout: element e owns rows 5e..5e+4.
  for (int r = lane; r < ROWS; r += 32) {
    int e = r / 5, j = r % 5;
    int src = (j < 4) ? (e * 4 + j) : (64 + e);
#pragma unroll 4
    for (int d = 0; d < 32; ++d) Z[r * 32 + d] = K[src * 32 + d];
  }
  __syncthreads();

  const float scale = 0.1767766952966369f; // 1/sqrt(32)

  // ---- two CTM blocks: fast (2 ticks) then slow (3 ticks) -----------------
  for (int blk = 0; blk < 2; ++blk) {
    const int nt = blk ? 3 : 2;
    const _Float16* wB = wt + (4 + blk * 7) * 1024;
    const float* ln_g = P.ln_g[blk];
    const float* ln_b = P.ln_b[blk];
    const float wPl = 1.0f / ((float)nt * (float)BATCH * 160.0f);
    const float wEp = 0.5f / ((float)nt * (float)BATCH);

    for (int t = 0; t < nt; ++t) {
      // h = LN(lin(z))
      lin5(Z, Pr, wB + 0 * 1024, P.bb[blk][0], false);
      __syncthreads();
      for (int r = lane; r < ROWS; r += 32) {
        const _Float16* p = &Pr[r * 32];
        float mu = 0.f;
#pragma unroll
        for (int d = 0; d < 32; ++d) mu += (float)p[d];
        mu *= (1.0f / 32.0f);
        float var = 0.f;
#pragma unroll
        for (int d = 0; d < 32; ++d) { float q = (float)p[d] - mu; var += q * q; }
        var *= (1.0f / 32.0f);
        float rs = rsqrtf(var + 1e-5f);
#pragma unroll
        for (int d = 0; d < 32; ++d)
          H[r * 32 + d] = (_Float16)(((float)p[d] - mu) * rs * ln_g[d] + ln_b[d]);
      }
      __syncthreads();

      lin5(H, Q, wB + 1 * 1024, P.bb[blk][1], false);
      lin5(H, K, wB + 2 * 1024, P.bb[blk][2], false);
      lin5(H, V, wB + 3 * 1024, P.bb[blk][3], false);
      __syncthreads();

      // attention phase A: sim rows + softmax row-max (= 1/sum(exp(s-m)))
      for (int p = lane; p < ROWS; p += 32) {
        int e = p / 5, q = p % 5;
        const _Float16* Qr = &Q[(5 * e + q) * 32];
        float qv[32];
#pragma unroll
        for (int d = 0; d < 32; ++d) qv[d] = (float)Qr[d];
        float s[5];
        for (int k = 0; k < 5; ++k) {
          const _Float16* Kr = &K[(5 * e + k) * 32];
          float acc = 0.f;
#pragma unroll
          for (int d = 0; d < 32; ++d) acc += qv[d] * (float)Kr[d];
          s[k] = acc * scale;
          SIM[e * 30 + q * 5 + k] = s[k];
        }
        float m = s[0];
        for (int k = 1; k < 5; ++k) m = fmaxf(m, s[k]);
        float sum = 0.f;
        for (int k = 0; k < 5; ++k) sum += expf(s[k] - m);
        SIM[e * 30 + 25 + q] = 1.0f / sum;   // max prob of softmax row
      }
      __syncthreads();

      // phase B: confidence -> eff_p per element
      for (int e = lane; e < 16; e += 32) {
        float mr = 0.f;
        for (int q = 0; q < 5; ++q) mr += SIM[e * 30 + 25 + q];
        mr *= 0.2f;
        float conf = (mr - 0.2f) / (0.8f + 1e-6f);
        conf = fminf(fmaxf(conf, 0.f), 1.f);
        float effp = 2.0f + 2.0f * conf;     // P_MIN + (P_MAX-P_MIN)*conf
        epAcc += wEp * effp;
        SIM[e * 30 + 25] = effp;             // row-max slots no longer needed
      }
      __syncthreads();

      // phase C: sharp softmax + context (overwrite Q rows; store attn in SIM)
      for (int p = lane; p < ROWS; p += 32) {
        int e = p / 5, q = p % 5;
        float eff = SIM[e * 30 + 25];
        float s[5];
        for (int k = 0; k < 5; ++k) s[k] = SIM[e * 30 + q * 5 + k] * eff;
        float m = s[0];
        for (int k = 1; k < 5; ++k) m = fmaxf(m, s[k]);
        float sum = 0.f;
        float a[5];
        for (int k = 0; k < 5; ++k) { a[k] = expf(s[k] - m); sum += a[k]; }
        float inv = 1.0f / sum;
        for (int k = 0; k < 5; ++k) { a[k] *= inv; SIM[e * 30 + q * 5 + k] = a[k]; }
        // context: key-outer, dim-inner so V rows stream contiguously
        float acc[32];
#pragma unroll
        for (int d = 0; d < 32; ++d) acc[d] = 0.f;
        for (int k = 0; k < 5; ++k) {
          float ak = a[k];
          const _Float16* Vr = &V[(5 * e + k) * 32];
#pragma unroll
          for (int d = 0; d < 32; ++d) acc[d] += ak * (float)Vr[d];
        }
        _Float16* Cr = &Q[(5 * e + q) * 32];
#pragma unroll
        for (int d = 0; d < 32; ++d) Cr[d] = (_Float16)acc[d];
      }
      __syncthreads();

      // target = o(ctx); z_pred = p2(relu(p1(z)))
      lin5(Q, V, wB + 4 * 1024, P.bb[blk][4], false);      // V := target
      lin5(Z, Pr, wB + 5 * 1024, P.bb[blk][5], true);
      __syncthreads();
      lin5(Pr, K, wB + 6 * 1024, P.bb[blk][6], false);     // K := z_pred
      __syncthreads();

      // z update + prediction loss (pure FMA chain, tau precomputed)
      for (int r = lane; r < ROWS; r += 32) {
        _Float16* zr = &Z[r * 32];
        const _Float16* tr = &V[r * 32];
        const _Float16* pr = &K[r * 32];
#pragma unroll
        for (int d = 0; d < 32; ++d) {
          float it = (float)sItau[blk][d];
          float z  = (float)zr[d];
          float zn = z + ((float)tr[d] - z) * it;
          float df = (float)pr[d] - zn;
          plAcc += wPl * df * df;
          zr[d] = (_Float16)zn;
        }
      }
      __syncthreads();
    }
  }

  // ---- head: slot logits -> log-softmax, mix by last-tick attn(q=4,k<4) ---
  for (int e = lane; e < 16; e += 32) {
    float logp[4][4];
    for (int s = 0; s < 4; ++s) {
      const _Float16* zr = &Z[(5 * e + s) * 32];
      float lg[4];
      for (int c = 0; c < 4; ++c) lg[c] = P.head_b[c];
#pragma unroll
      for (int d = 0; d < 32; ++d) {
        float zv = (float)zr[d];
        for (int c = 0; c < 4; ++c) lg[c] += zv * P.head_w[d * 4 + c];
      }
      float m = fmaxf(fmaxf(lg[0], lg[1]), fmaxf(lg[2], lg[3]));
      float sum = 0.f;
      for (int c = 0; c < 4; ++c) sum += expf(lg[c] - m);
      float lse = m + logf(sum);
      for (int c = 0; c < 4; ++c) logp[s][c] = lg[c] - lse;
    }
    float a[4], asum = 1e-8f;
    for (int k = 0; k < 4; ++k) { a[k] = SIM[e * 30 + 20 + k]; asum += a[k]; }
    for (int k = 0; k < 4; ++k) a[k] = logf(fmaxf(a[k] / asum, 1e-8f));
    for (int c = 0; c < 4; ++c) {
      float m = logp[0][c] + a[0];
      for (int s = 1; s < 4; ++s) m = fmaxf(m, logp[s][c] + a[s]);
      float sum = 0.f;
      for (int s = 0; s < 4; ++s) sum += expf(logp[s][c] + a[s] - m);
      P.out[(eBase + e) * 4 + c] = m + logf(sum);
    }
  }

  // ---- wave32 reduction of the two scalars, then global f32 atomics -------
  float pl = plAcc, ep = epAcc;
#pragma unroll
  for (int off = 16; off; off >>= 1) {
    pl += __shfl_xor(pl, off, 32);
    ep += __shfl_xor(ep, off, 32);
  }
  if (lane == 0) {
    atomicAdd(P.out + SCALAR_BASE,     pl);
    atomicAdd(P.out + SCALAR_BASE + 1, ep);
  }
}

// ---------------------------------------------------------------------------

extern "C" void kernel_launch(void* const* d_in, const int* in_sizes, int n_in,
                              void* d_out, int out_size, void* d_ws, size_t ws_size,
                              hipStream_t stream) {
  const float* in[45];
  for (int i = 0; i < 45 && i < n_in; ++i) in[i] = (const float*)d_in[i];
  _Float16* wt = (_Float16*)d_ws;
  float* out = (float*)d_out;

  // weights -> transposed f16 tiles in d_ws (order: se1,se2,qe1,qe2,
  // fast{in,q,k,v,o,p1,p2}, slow{...})
  PrepP pp;
  const int wsrc[18] = {1, 3, 5, 7,  9, 13, 15, 17, 19, 21, 23,
                        26, 30, 32, 34, 36, 38, 40};
  for (int m = 0; m < 18; ++m) pp.w[m] = in[wsrc[m]];
  prep_weights<<<72, 256, 0, stream>>>(pp, wt);
  init_out<<<1, 64, 0, stream>>>(out);

  MainP mp;
  mp.x = in[0];
  mp.out = out;
  mp.eb[0] = in[2];  mp.eb[1] = in[4];  mp.eb[2] = in[6];  mp.eb[3] = in[8];
  const int fb[7] = {10, 14, 16, 18, 20, 22, 24};
  for (int i = 0; i < 7; ++i) { mp.bb[0][i] = in[fb[i]]; mp.bb[1][i] = in[fb[i] + 17]; }
  mp.ln_g[0] = in[11]; mp.ln_b[0] = in[12]; mp.ltau[0] = in[25];
  mp.ln_g[1] = in[28]; mp.ln_b[1] = in[29]; mp.ltau[1] = in[42];
  mp.head_w = in[43];  mp.head_b = in[44];

  const int nBlocks = BATCH / (ELEM_PER_WAVE * WAVES_PER_BLOCK); // 32768
  ctm_kernel<<<nBlocks, 32 * WAVES_PER_BLOCK, 0, stream>>>(mp, wt);
}
